// TexVisFusion_8211977470294
// MI455X (gfx1250) — compile-verified, matrix operand order checked by reading, not run on hardware
//
#include <hip/hip_runtime.h>
#include <hip/hip_bf16.h>
#include <math.h>

#define NQ    65536
#define NVERT 1558
#define NUMV_ 779

// Packed-weight sub-offsets (in halfs) inside the contiguous weight block.
#define W_AT1 0          // 6 Mtiles * 3 Ktiles * 512 = 9216
#define W_AT2 9216       // 1 * 3 * 512 = 1536
#define W_F1  10752      // 6 * 3 * 512 = 9216
#define W_F2  19968      // 3 * 3 * 512 = 4608
#define W_TOT 24576      // halfs (= 49152 bytes)

typedef __attribute__((ext_vector_type(16))) _Float16 v16h;
typedef __attribute__((ext_vector_type(8)))  float    v8f;

__device__ __forceinline__ v8f wmma16(v16h a, v16h b, v8f c) {
  // D = A(16x32 f16) * B(32x16 f16) + C(16x16 f32)
  return __builtin_amdgcn_wmma_f32_16x16x32_f16(false, a, false, b, (short)0, c,
                                                false, false);
}

// ---------------------------------------------------------------------------
// Bilinear feature sampling at vertex positions: vf[:, 0:3] from img_fmap
// (3x256x256), vf[:, 3:11] from ft1 (8x64x64). vf row stride = 29.
// ---------------------------------------------------------------------------
__global__ void k_feat_sample(const float* __restrict__ vert_xy,
                              const float* __restrict__ img_fmap,
                              const float* __restrict__ ft1,
                              float* __restrict__ vf) {
  int vtx = blockIdx.x * blockDim.x + threadIdx.x;
  if (vtx >= NVERT) return;
  float sx = vert_xy[vtx * 2 + 0], sy = vert_xy[vtx * 2 + 1];
  {
    const int W = 256, H = 256;
    float x = (sx + 1.f) * 0.5f * (W - 1), y = (sy + 1.f) * 0.5f * (H - 1);
    float x0f = floorf(x), y0f = floorf(y);
    float wx = x - x0f, wy = y - y0f;
    int x0 = min(max((int)x0f, 0), W - 1), x1 = min(max((int)x0f + 1, 0), W - 1);
    int y0 = min(max((int)y0f, 0), H - 1), y1 = min(max((int)y0f + 1, 0), H - 1);
    for (int c = 0; c < 3; ++c) {
      const float* f = img_fmap + c * H * W;
      float v00 = f[y0 * W + x0], v01 = f[y0 * W + x1];
      float v10 = f[y1 * W + x0], v11 = f[y1 * W + x1];
      vf[vtx * 29 + c] = v00 * (1.f - wx) * (1.f - wy) + v01 * wx * (1.f - wy) +
                         v10 * (1.f - wx) * wy + v11 * wx * wy;
    }
  }
  {
    const int W = 64, H = 64;
    float x = (sx + 1.f) * 0.5f * (W - 1), y = (sy + 1.f) * 0.5f * (H - 1);
    float x0f = floorf(x), y0f = floorf(y);
    float wx = x - x0f, wy = y - y0f;
    int x0 = min(max((int)x0f, 0), W - 1), x1 = min(max((int)x0f + 1, 0), W - 1);
    int y0 = min(max((int)y0f, 0), H - 1), y1 = min(max((int)y0f + 1, 0), H - 1);
    for (int c = 0; c < 8; ++c) {
      const float* f = ft1 + c * H * W;
      float v00 = f[y0 * W + x0], v01 = f[y0 * W + x1];
      float v10 = f[y1 * W + x0], v11 = f[y1 * W + x1];
      vf[vtx * 29 + 3 + c] = v00 * (1.f - wx) * (1.f - wy) + v01 * wx * (1.f - wy) +
                             v10 * (1.f - wx) * wy + v11 * wx * wy;
    }
  }
}

// ---------------------------------------------------------------------------
// conv2d 3x3 pad1 + LayerNorm over (H,W) per channel (gamma/beta per pixel)
// + relu. One block per output channel.
// ---------------------------------------------------------------------------
__global__ __launch_bounds__(256) void k_conv3x3_ln_relu(
    const float* __restrict__ in, const float* __restrict__ w,
    const float* __restrict__ g, const float* __restrict__ b,
    float* __restrict__ out, int Cin, int H, int W) {
  int c = blockIdx.x;
  int HW = H * W;
  const float* wc = w + c * Cin * 9;
  float s1 = 0.f, s2 = 0.f;
  for (int p = threadIdx.x; p < HW; p += 256) {
    int y = p / W, x = p - y * W;
    float acc = 0.f;
    for (int ic = 0; ic < Cin; ++ic) {
      const float* fin = in + ic * HW;
      const float* wk = wc + ic * 9;
      for (int ky = 0; ky < 3; ++ky) {
        int iy = y + ky - 1;
        if (iy < 0 || iy >= H) continue;
        for (int kx = 0; kx < 3; ++kx) {
          int ix = x + kx - 1;
          if (ix < 0 || ix >= W) continue;
          acc += fin[iy * W + ix] * wk[ky * 3 + kx];
        }
      }
    }
    out[c * HW + p] = acc;
    s1 += acc;
    s2 += acc * acc;
  }
  __shared__ float r1[256], r2[256];
  r1[threadIdx.x] = s1;
  r2[threadIdx.x] = s2;
  __syncthreads();
  for (int s = 128; s > 0; s >>= 1) {
    if (threadIdx.x < s) {
      r1[threadIdx.x] += r1[threadIdx.x + s];
      r2[threadIdx.x] += r2[threadIdx.x + s];
    }
    __syncthreads();
  }
  float mean = r1[0] / (float)HW;
  float var = r2[0] / (float)HW - mean * mean;
  float inv = rsqrtf(var + 1e-6f);
  for (int p = threadIdx.x; p < HW; p += 256) {
    float vv = (out[c * HW + p] - mean) * inv * g[p] + b[p];
    out[c * HW + p] = vv > 0.f ? vv : 0.f;
  }
}

// ---------------------------------------------------------------------------
// Adaptive 3x3 average pool (overlapping floor/ceil bins), writes gf[c][colOff+cell].
// Grid: 42*9 blocks of 64 threads.
// ---------------------------------------------------------------------------
__global__ void k_apool3(const float* __restrict__ in, int H, int W,
                         float* __restrict__ gf, int colOff) {
  int c = blockIdx.x / 9, cell = blockIdx.x % 9;
  int hi = cell / 3, wi = cell % 3;
  int hs = (hi * H) / 3, he = ((hi + 1) * H + 2) / 3;
  int wss = (wi * W) / 3, we = ((wi + 1) * W + 2) / 3;
  const float* f = in + c * H * W;
  int ww = we - wss;
  int cnt = (he - hs) * ww;
  float s = 0.f;
  for (int t = threadIdx.x; t < cnt; t += 64) {
    int r = t / ww, col = t - r * ww;
    s += f[(hs + r) * W + wss + col];
  }
  __shared__ float red[64];
  red[threadIdx.x] = s;
  __syncthreads();
  for (int st = 32; st > 0; st >>= 1) {
    if (threadIdx.x < st) red[threadIdx.x] += red[threadIdx.x + st];
    __syncthreads();
  }
  if (threadIdx.x == 0) gf[c * 18 + colOff + cell] = red[0] / (float)cnt;
}

// ---------------------------------------------------------------------------
// conv1d k=3 pad1 over L=18 + LayerNorm over L (gamma/beta per position) + relu.
// One block (32 threads) per output channel; writes out[c*outStride+outOff+t].
// ---------------------------------------------------------------------------
__global__ void k_conv1d3_ln_relu(const float* __restrict__ in,
                                  const float* __restrict__ w,
                                  const float* __restrict__ g,
                                  const float* __restrict__ b,
                                  float* __restrict__ out, int Cin,
                                  int outStride, int outOff) {
  const int L = 18;
  int c = blockIdx.x;
  __shared__ float vals[L];
  __shared__ float mv[2];
  int t = threadIdx.x;
  if (t < L) {
    float acc = 0.f;
    const float* wc = w + c * Cin * 3;
    for (int ic = 0; ic < Cin; ++ic) {
      const float* fi = in + ic * L;
      const float* wk = wc + ic * 3;
      for (int k = 0; k < 3; ++k) {
        int p = t + k - 1;
        if (p >= 0 && p < L) acc += fi[p] * wk[k];
      }
    }
    vals[t] = acc;
  }
  __syncthreads();
  if (t == 0) {
    float m = 0.f;
    for (int i2 = 0; i2 < L; ++i2) m += vals[i2];
    m /= (float)L;
    float v2 = 0.f;
    for (int i2 = 0; i2 < L; ++i2) {
      float d = vals[i2] - m;
      v2 += d * d;
    }
    v2 /= (float)L;
    mv[0] = m;
    mv[1] = rsqrtf(v2 + 1e-6f);
  }
  __syncthreads();
  if (t < L) {
    float vv = (vals[t] - mv[0]) * mv[1] * g[t] + b[t];
    out[c * outStride + outOff + t] = vv > 0.f ? vv : 0.f;
  }
}

// ---------------------------------------------------------------------------
// Nearest-vertex argmin: one thread per query, vertices cached in LDS.
// ---------------------------------------------------------------------------
__global__ __launch_bounds__(256) void k_argmin(const float* __restrict__ v,
                                                const float* __restrict__ vert,
                                                int* __restrict__ idx) {
  __shared__ float pv[NVERT * 3];
  for (int i = threadIdx.x; i < NVERT * 3; i += 256) pv[i] = vert[i];
  __syncthreads();
  int n = blockIdx.x * 256 + threadIdx.x;
  float qx = v[n * 3 + 0], qy = v[n * 3 + 1], qz = v[n * 3 + 2];
  float best = 3.4e38f;
  int bi = 0;
  for (int p = 0; p < NVERT; ++p) {
    float dx = qx - pv[p * 3 + 0];
    float dy = qy - pv[p * 3 + 1];
    float dz = qz - pv[p * 3 + 2];
    float d = dx * dx + dy * dy + dz * dz;
    if (d < best) { best = d; bi = p; }
  }
  idx[n] = bi;
}

// ---------------------------------------------------------------------------
// Pack an OxC(=96) f32 weight matrix into f16 WMMA A-fragments
// (ISA 7.12.2 16-bit A layout). dst index = ((mt*3+kt)*32+lane)*16+h.
// Rows o >= Oreal are zero-padded.
// ---------------------------------------------------------------------------
__global__ void k_pack_w(const float* __restrict__ W, _Float16* __restrict__ dst,
                         int Oreal, int Mtiles) {
  int tid = blockIdx.x * blockDim.x + threadIdx.x;
  int total = Mtiles * 3 * 512;
  if (tid >= total) return;
  int h = tid & 15;
  int lane = (tid >> 4) & 31;
  int rest = tid >> 9;  // mt*3+kt
  int o = (rest / 3) * 16 + (lane & 15);
  int koff = (h & 7) | ((h >> 3) << 4) | ((lane >> 4) << 3);
  int cc = (rest % 3) * 32 + koff;
  float val = (o < Oreal) ? W[o * 96 + cc] : 0.f;
  dst[tid] = (_Float16)val;
}

// ---------------------------------------------------------------------------
// Fused per-query MLP: gather y(96) -> h1=relu(at_w1@y) -> at=sigmoid(at_w2@h1)
// -> y2 gated -> h2=relu(fconv_w1@y2) -> out=fconv_w2@h2 (40).
// One wave = 16 queries; block = 8 waves (256 thr); grid = NQ/128.
// Packed f16 weights (48KB) are staged once per block into LDS using the
// CDNA5 async Global->LDS path (GLOBAL_LOAD_ASYNC_TO_LDS_B128 / ASYNCcnt),
// then all A-fragments come from LDS. 48 v_wmma_f32_16x16x32_f16 per wave.
// ---------------------------------------------------------------------------
__global__ __launch_bounds__(256) void k_fused_mlp(
    const float* __restrict__ ft_xy, const float* __restrict__ query_vis,
    const float* __restrict__ img_xy, const float* __restrict__ latent,
    const float* __restrict__ vf, const float* __restrict__ vert_vis,
    const int* __restrict__ idx, const _Float16* __restrict__ pw_all,
    float* __restrict__ out) {
  __shared__ __align__(32) _Float16 wlds[W_TOT];    // staged weights (48KB)
  __shared__ __align__(32) _Float16 ybuf[8][1536];  // per-wave B-frag-packed y
  __shared__ __align__(32) _Float16 hbuf[8][1536];  // per-wave B-frag-packed h
  __shared__ float atb[8][6][16];

  int tid = threadIdx.x, wave = tid >> 5, lane = tid & 31;
  int q = lane & 15, laneHi = lane >> 4;
  int n = (blockIdx.x * 8 + wave) * 16 + q;

  // --- Async stage of packed weights into LDS (3072 x 16B chunks) ---------
  {
    unsigned ldsBase = (unsigned)(size_t)(&wlds[0]);
    const char* gw = (const char*)pw_all;
#pragma unroll
    for (int it = 0; it < 12; ++it) {
      int chunk = it * 256 + tid;
      unsigned laddr = ldsBase + (unsigned)(chunk * 16);
      const char* ga = gw + chunk * 16;
      asm volatile("global_load_async_to_lds_b128 %0, %1, off"
                   :: "v"(laddr), "v"(ga)
                   : "memory");
    }
  }

  // --- Gather y[96] for this query (overlaps with async weight DMA) -------
  int j = idx[n];
  int jt = j + NUMV_;
  if (jt >= NVERT) jt -= NVERT;
  float visj = vert_vis[j], visjt = vert_vis[jt];
  const float* vfj = vf + j * 29;
  const float* vfjt = vf + jt * 29;

  // lanes 0-15: channels 0-47 ; lanes 16-31: channels 48-95.
  // B-fragment-packed store: for channel c, query q:
  //   kt=c/32, koff=c%32, lds_lane=q+16*((koff>>3)&1),
  //   lds_h=(koff&7)|(((koff>>4)&1)<<3)
  for (int ci = 0; ci < 48; ++ci) {
    int c = laneHi * 48 + ci;
    float val;
    if (c < 3)        val = img_xy[n * 3 + c];
    else if (c < 11)  val = ft_xy[n * 8 + (c - 3)];
    else if (c < 22)  val = vfj[c - 11] * visj;          // knn_f
    else if (c < 33)  val = vfjt[c - 22] * visjt;        // knn_toh_f
    else if (c < 51)  val = vfj[11 + (c - 33)] * visj;   // knn_gf
    else if (c < 69)  val = vfjt[11 + (c - 51)] * visjt; // knn_toh_gf
    else if (c < 93)  val = latent[n * 24 + (c - 69)];
    else if (c == 93) val = query_vis[n];
    else if (c == 94) val = visj;
    else              val = visjt;
    int kt = c >> 5, koff = c & 31;
    int bl = q + (((koff >> 3) & 1) << 4);
    int bh = (koff & 7) | (((koff >> 4) & 1) << 3);
    ybuf[wave][(kt * 32 + bl) * 16 + bh] = (_Float16)val;
  }
  asm volatile("s_wait_asynccnt 0x0" ::: "memory");
  __syncthreads();

  const v16h* wa1 = (const v16h*)&wlds[W_AT1];
  const v16h* wa2 = (const v16h*)&wlds[W_AT2];
  const v16h* wf1 = (const v16h*)&wlds[W_F1];
  const v16h* wf2 = (const v16h*)&wlds[W_F2];

  v16h yb[3];
#pragma unroll
  for (int kt = 0; kt < 3; ++kt)
    yb[kt] = ((const v16h*)&ybuf[wave][0])[kt * 32 + lane];

  const v8f vzero = {0.f, 0.f, 0.f, 0.f, 0.f, 0.f, 0.f, 0.f};

  // GEMM1: h1 = relu(at_w1 @ y), 96x96 * 96x16
#pragma unroll
  for (int mt = 0; mt < 6; ++mt) {
    v8f acc = vzero;
#pragma unroll
    for (int kt = 0; kt < 3; ++kt)
      acc = wmma16(wa1[(mt * 3 + kt) * 32 + lane], yb[kt], acc);
#pragma unroll
    for (int r = 0; r < 8; ++r) {
      float hv = acc[r];
      hv = hv > 0.f ? hv : 0.f;
      int c = mt * 16 + r + (laneHi << 3);
      int kt2 = c >> 5, koff = c & 31;
      int bl = q + (((koff >> 3) & 1) << 4);
      int bh = (koff & 7) | (((koff >> 4) & 1) << 3);
      hbuf[wave][(kt2 * 32 + bl) * 16 + bh] = (_Float16)hv;
    }
  }
  __syncthreads();

  v16h hb[3];
#pragma unroll
  for (int kt = 0; kt < 3; ++kt)
    hb[kt] = ((const v16h*)&hbuf[wave][0])[kt * 32 + lane];

  // GEMM2: at = sigmoid(at_w2 @ h1) (rows 0..5 of padded 16x96)
  {
    v8f acc = vzero;
#pragma unroll
    for (int kt = 0; kt < 3; ++kt)
      acc = wmma16(wa2[kt * 32 + lane], hb[kt], acc);
    if (laneHi == 0) {
#pragma unroll
      for (int r = 0; r < 6; ++r)
        atb[wave][r][q] = 1.f / (1.f + __expf(-acc[r]));
    }
  }
  __syncthreads();

  float gate[6];
#pragma unroll
  for (int s = 0; s < 6; ++s) gate[s] = atb[wave][s][q];

  // Build gated y2 fragments in registers.
  v16h y2[3];
#pragma unroll
  for (int kt = 0; kt < 3; ++kt) {
#pragma unroll
    for (int h = 0; h < 16; ++h) {
      int koff = (h & 7) | ((h >> 3) << 4) | (laneHi << 3);
      int c = kt * 32 + koff;
      float gv = (c < 11) ? gate[0]
                 : (c < 22) ? gate[1]
                 : (c < 33) ? gate[2]
                 : (c < 51) ? gate[3]
                 : (c < 69) ? gate[4]
                 : (c < 93) ? gate[5]
                            : 1.f;
      y2[kt][h] = (_Float16)((float)yb[kt][h] * gv);
    }
  }

  // GEMM3: h2 = relu(fconv_w1 @ y2)
#pragma unroll
  for (int mt = 0; mt < 6; ++mt) {
    v8f acc = vzero;
#pragma unroll
    for (int kt = 0; kt < 3; ++kt)
      acc = wmma16(wf1[(mt * 3 + kt) * 32 + lane], y2[kt], acc);
#pragma unroll
    for (int r = 0; r < 8; ++r) {
      float hv = acc[r];
      hv = hv > 0.f ? hv : 0.f;
      int c = mt * 16 + r + (laneHi << 3);
      int kt2 = c >> 5, koff = c & 31;
      int bl = q + (((koff >> 3) & 1) << 4);
      int bh = (koff & 7) | (((koff >> 4) & 1) << 3);
      hbuf[wave][(kt2 * 32 + bl) * 16 + bh] = (_Float16)hv;
    }
  }
  __syncthreads();

  v16h h2[3];
#pragma unroll
  for (int kt = 0; kt < 3; ++kt)
    h2[kt] = ((const v16h*)&hbuf[wave][0])[kt * 32 + lane];

  // GEMM4: out = fconv_w2 @ h2 (rows 0..39 of padded 48x96)
#pragma unroll
  for (int mt = 0; mt < 3; ++mt) {
    v8f acc = vzero;
#pragma unroll
    for (int kt = 0; kt < 3; ++kt)
      acc = wmma16(wf2[(mt * 3 + kt) * 32 + lane], h2[kt], acc);
#pragma unroll
    for (int r = 0; r < 8; ++r) {
      int o = mt * 16 + r + (laneHi << 3);
      if (o < 40) out[(size_t)n * 40 + o] = acc[r];
    }
  }
}

// ---------------------------------------------------------------------------
extern "C" void kernel_launch(void* const* d_in, const int* in_sizes, int n_in,
                              void* d_out, int out_size, void* d_ws,
                              size_t ws_size, hipStream_t stream) {
  (void)in_sizes; (void)n_in; (void)out_size; (void)ws_size;
  const float* vert_xy   = (const float*)d_in[0];
  const float* ft1       = (const float*)d_in[1];
  const float* ft_xy     = (const float*)d_in[2];
  const float* vert      = (const float*)d_in[3];
  const float* v         = (const float*)d_in[4];
  const float* vert_vis  = (const float*)d_in[5];
  const float* query_vis = (const float*)d_in[6];
  const float* img_xy    = (const float*)d_in[7];
  const float* img_fmap  = (const float*)d_in[8];
  const float* latent    = (const float*)d_in[9];
  const float* fconv_w1  = (const float*)d_in[10];
  const float* fconv_w2  = (const float*)d_in[11];
  const float* at_w1     = (const float*)d_in[12];
  const float* at_w2     = (const float*)d_in[13];
  const float* gt_w1     = (const float*)d_in[14];
  const float* gt_g1     = (const float*)d_in[15];
  const float* gt_b1     = (const float*)d_in[16];
  const float* gt_w2     = (const float*)d_in[17];
  const float* gt_g2     = (const float*)d_in[18];
  const float* gt_b2     = (const float*)d_in[19];
  const float* c3_w1     = (const float*)d_in[20];
  const float* c3_g1     = (const float*)d_in[21];
  const float* c3_b1     = (const float*)d_in[22];
  const float* c3_w2     = (const float*)d_in[23];
  const float* c3_g2     = (const float*)d_in[24];
  const float* c3_b2     = (const float*)d_in[25];
  const float* c4_w1     = (const float*)d_in[26];
  const float* c4_g1     = (const float*)d_in[27];
  const float* c4_b1     = (const float*)d_in[28];
  const float* c4_w2     = (const float*)d_in[29];
  const float* c4_g2     = (const float*)d_in[30];
  const float* c4_b2     = (const float*)d_in[31];
  float* out = (float*)d_out;

  char* base = (char*)d_ws;
  size_t off = 0;
  auto carve = [&](size_t bytes) -> char* {
    char* p = base + off;
    off = (off + bytes + 255) & ~(size_t)255;
    return p;
  };
  float*    vf     = (float*)carve((size_t)NVERT * 29 * sizeof(float));
  int*      idx    = (int*)carve((size_t)NQ * sizeof(int));
  float*    gf     = (float*)carve(42 * 18 * sizeof(float));
  float*    hgt1   = (float*)carve(779 * 18 * sizeof(float));
  _Float16* pw_all = (_Float16*)carve((size_t)W_TOT * sizeof(_Float16));
  float*    tmpA   = (float*)carve((size_t)21 * 256 * 256 * sizeof(float));
  float*    tmpB   = (float*)carve((size_t)42 * 256 * 256 * sizeof(float));

  // Weight packing (f32 -> f16 WMMA fragments, one contiguous block)
  k_pack_w<<<(6 * 3 * 512 + 255) / 256, 256, 0, stream>>>(at_w1, pw_all + W_AT1, 96, 6);
  k_pack_w<<<(1 * 3 * 512 + 255) / 256, 256, 0, stream>>>(at_w2, pw_all + W_AT2, 6, 1);
  k_pack_w<<<(6 * 3 * 512 + 255) / 256, 256, 0, stream>>>(fconv_w1, pw_all + W_F1, 96, 6);
  k_pack_w<<<(3 * 3 * 512 + 255) / 256, 256, 0, stream>>>(fconv_w2, pw_all + W_F2, 40, 3);

  // Vertex feature sampling -> vf[:, 0:11]
  k_feat_sample<<<(NVERT + 255) / 256, 256, 0, stream>>>(vert_xy, img_fmap, ft1, vf);

  // c3 stack (64x64), pooled into gf[:, 9:18]
  k_conv3x3_ln_relu<<<21, 256, 0, stream>>>(ft1, c3_w1, c3_g1, c3_b1, tmpA, 8, 64, 64);
  k_conv3x3_ln_relu<<<42, 256, 0, stream>>>(tmpA, c3_w2, c3_g2, c3_b2, tmpB, 21, 64, 64);
  k_apool3<<<42 * 9, 64, 0, stream>>>(tmpB, 64, 64, gf, 9);

  // c4 stack (256x256), pooled into gf[:, 0:9]
  k_conv3x3_ln_relu<<<21, 256, 0, stream>>>(img_fmap, c4_w1, c4_g1, c4_b1, tmpA, 3, 256, 256);
  k_conv3x3_ln_relu<<<42, 256, 0, stream>>>(tmpA, c4_w2, c4_g2, c4_b2, tmpB, 21, 256, 256);
  k_apool3<<<42 * 9, 64, 0, stream>>>(tmpB, 256, 256, gf, 0);

  // gt stack: gf(42x18) -> hgt1(779x18) -> vf[:, 11:29] (1558x18)
  k_conv1d3_ln_relu<<<779, 32, 0, stream>>>(gf, gt_w1, gt_g1, gt_b1, hgt1, 42, 18, 0);
  k_conv1d3_ln_relu<<<1558, 32, 0, stream>>>(hgt1, gt_w2, gt_g2, gt_b2, vf, 779, 29, 11);

  // Nearest vertex per query
  k_argmin<<<NQ / 256, 256, 0, stream>>>(v, vert, idx);

  // Fused WMMA MLP -> out (65536 x 40)
  k_fused_mlp<<<NQ / 128, 256, 0, stream>>>(ft_xy, query_vis, img_xy, latent, vf,
                                            vert_vis, idx, pw_all, out);
}
